// ATSBlock_62886911148417
// MI455X (gfx1250) — matrix operand order repaired and checked by reading, not compiled, not run on hardware
//
#include <hip/hip_runtime.h>
#include <math.h>

#define SS   1025
#define BB   4
#define DD   1024
#define HH   16
#define HD   64
#define TD   3072      // 3*D
#define NT   512       // n_tokens
#define MLPD 4096
#define SP   1056      // S padded to multiple of 32
#define EPSF 1e-6f

typedef __bf16 bf16;
typedef bf16  v16bf __attribute__((ext_vector_type(16)));
typedef float v8f   __attribute__((ext_vector_type(8)));

__device__ inline v8f wmma_bf16(v16bf a, v16bf b, v8f c) {
  return __builtin_amdgcn_wmma_f32_16x16x32_bf16(false, a, false, b, (short)0, c, false, false);
}

// A fragment: 16x32 bf16, row m per lane (m = lane&15), contiguous K:
// elems 0..15 -> k = half*16 + e, half = lane>>4
__device__ inline v16bf load_a_frag(const bf16* __restrict__ A, long lda, int m, int k0, int lane) {
  const bf16* p = A + (long)m * lda + k0 + ((lane >> 4) << 4);
  v16bf r;
#pragma unroll
  for (int i = 0; i < 8; ++i) { r[i] = p[i]; r[8 + i] = p[8 + i]; }
  return r;
}

// B fragment: 32x16 bf16 from Bt stored [N,K] row-major. lane column n, k = k0 + (lane>>4)*16, 16 contiguous.
__device__ inline v16bf load_b_frag(const bf16* __restrict__ Bt, long ldb, int n, int k0, int lane) {
  const bf16* p = Bt + (long)n * ldb + k0 + ((lane >> 4) << 4);
  v16bf r;
#pragma unroll
  for (int i = 0; i < 16; ++i) r[i] = p[i];
  return r;
}

__device__ inline v16bf zero_frag() {
  v16bf r;
#pragma unroll
  for (int i = 0; i < 16; ++i) r[i] = (bf16)0.0f;
  return r;
}

// ---------------- LayerNorm 1 ----------------
__global__ void ln1_kernel(const float* __restrict__ x, const float* __restrict__ g,
                           const float* __restrict__ bia, float* __restrict__ yf,
                           bf16* __restrict__ yb) {
  int row = blockIdx.x;                       // token row = s*B + b
  const float* xr = x + (long)row * DD;
  __shared__ float red[256];
  int tid = threadIdx.x;
  float v[4], s = 0.f, s2 = 0.f;
#pragma unroll
  for (int i = 0; i < 4; ++i) { v[i] = xr[tid + i * 256]; s += v[i]; s2 += v[i] * v[i]; }
  red[tid] = s; __syncthreads();
  for (int st = 128; st > 0; st >>= 1) { if (tid < st) red[tid] += red[tid + st]; __syncthreads(); }
  float mu = red[0] / DD; __syncthreads();
  red[tid] = s2; __syncthreads();
  for (int st = 128; st > 0; st >>= 1) { if (tid < st) red[tid] += red[tid + st]; __syncthreads(); }
  float var = red[0] / DD - mu * mu;
  float rs = rsqrtf(var + EPSF);
#pragma unroll
  for (int i = 0; i < 4; ++i) {
    int d = tid + i * 256;
    float o = (v[i] - mu) * rs * g[d] + bia[d];
    yf[(long)row * DD + d] = o;
    yb[(long)row * DD + d] = (bf16)o;
  }
}

// ---------------- weight conversion ----------------
__global__ void cvt_kernel(const float* __restrict__ w, bf16* __restrict__ o, long n) {
  long i = (long)blockIdx.x * blockDim.x + threadIdx.x;
  if (i < n) o[i] = (bf16)w[i];
}
__global__ void transpose_cvt_kernel(const float* __restrict__ w, bf16* __restrict__ o, int K, int N) {
  long i = (long)blockIdx.x * blockDim.x + threadIdx.x;
  if (i < (long)K * N) { int k = (int)(i / N), n = (int)(i % N); o[(long)n * K + k] = (bf16)w[i]; }
}

// ---------------- generic GEMM: C[M,N] = A[M,K](bf16) * Bt[N,K]^T + bias ----------------
// 4 waves / block; each wave computes a 32x64 tile (2 A-frags x 4 B-frags = 8 WMMA per K-step).
__global__ void gemm_bt_kernel(const bf16* __restrict__ A, const bf16* __restrict__ Bt,
                               const float* __restrict__ bias, float* __restrict__ Cf,
                               bf16* __restrict__ Cb, int M, int N, int K, int act) {
  int lane = threadIdx.x & 31, wave = threadIdx.x >> 5;
  int m0 = blockIdx.y * 128 + wave * 32;
  int n0 = blockIdx.x * 64;
  v8f acc[2][4];
#pragma unroll
  for (int s = 0; s < 2; ++s)
#pragma unroll
    for (int t = 0; t < 4; ++t)
#pragma unroll
      for (int i = 0; i < 8; ++i) acc[s][t][i] = 0.f;
  int mA0 = m0 + (lane & 15);      if (mA0 >= M) mA0 = M - 1;
  int mA1 = m0 + 16 + (lane & 15); if (mA1 >= M) mA1 = M - 1;
  for (int k0 = 0; k0 < K; k0 += 32) {
    if (k0 + 32 < K) __builtin_prefetch(A + (long)mA0 * K + k0 + 32, 0, 1);
    v16bf a0 = load_a_frag(A, K, mA0, k0, lane);
    v16bf a1 = load_a_frag(A, K, mA1, k0, lane);
#pragma unroll
    for (int t = 0; t < 4; ++t) {
      v16bf b = load_b_frag(Bt, K, n0 + t * 16 + (lane & 15), k0, lane);
      acc[0][t] = wmma_bf16(a0, b, acc[0][t]);
      acc[1][t] = wmma_bf16(a1, b, acc[1][t]);
    }
  }
#pragma unroll
  for (int t = 0; t < 4; ++t) {
    int n = n0 + t * 16 + (lane & 15);
    float bv = bias ? bias[n] : 0.f;
#pragma unroll
    for (int s = 0; s < 2; ++s)
#pragma unroll
      for (int i = 0; i < 8; ++i) {
        int m = m0 + s * 16 + ((lane >> 4) << 3) + i;
        if (m < M) {
          float v = acc[s][t][i] + bv;
          if (act) v = 0.5f * v * (1.0f + erff(v * 0.70710678118654752f));   // exact GELU
          if (Cf) Cf[(long)m * N + n] = v;
          if (Cb) Cb[(long)m * N + n] = (bf16)v;
        }
      }
  }
}

// ---------------- attention row 0 (softmax with policy) ----------------
__global__ void attn0_kernel(const bf16* __restrict__ qkv, const float* __restrict__ pol,
                             float* __restrict__ attn0) {
  int bh = blockIdx.x, b = bh / HH, h = bh % HH;
  const bf16* q0 = qkv + (long)b * TD + h * HD;   // token s=0 -> row b
  __shared__ float red[256];
  int tid = threadIdx.x;
  float sc[5], mx = -1e30f;
#pragma unroll
  for (int i = 0; i < 5; ++i) {
    int t = tid + i * 256;
    float s = -1e30f;
    if (t < SS) {
      const bf16* kr = qkv + (long)(t * BB + b) * TD + DD + h * HD;
      float a = 0.f;
      for (int d = 0; d < HD; ++d) a += (float)q0[d] * (float)kr[d];
      s = a * 0.125f;                              // HD^-0.5
    }
    sc[i] = s; mx = fmaxf(mx, s);
  }
  red[tid] = mx; __syncthreads();
  for (int st = 128; st > 0; st >>= 1) { if (tid < st) red[tid] = fmaxf(red[tid], red[tid + st]); __syncthreads(); }
  float rmax = red[0]; __syncthreads();
  float sum = 0.f;
#pragma unroll
  for (int i = 0; i < 5; ++i) {
    int t = tid + i * 256;
    if (t < SS) {
      float ap = pol[b * SS + t];
      float apf = ap + (1.f - ap) * ((t == 0) ? 1.f : 0.f);  // eye term, query row 0
      float e = __expf(sc[i] - rmax) * apf;
      sc[i] = e; sum += e;
    }
  }
  red[tid] = sum; __syncthreads();
  for (int st = 128; st > 0; st >>= 1) { if (tid < st) red[tid] += red[tid + st]; __syncthreads(); }
  float inv = 1.f / (red[0] + EPSF);
#pragma unroll
  for (int i = 0; i < 5; ++i) {
    int t = tid + i * 256;
    if (t < SS) attn0[(long)bh * SS + t] = (sc[i] + EPSF / SS) * inv;
  }
}

// ---------------- v-norm * head-summed attention row-0 -> significance ----------------
__global__ void sig_kernel(const bf16* __restrict__ qkv, const float* __restrict__ attn0,
                           float* __restrict__ sig) {
  int id = blockIdx.x * blockDim.x + threadIdx.x;
  if (id >= BB * SS) return;
  int b = id / SS, t = id % SS;
  const bf16* vr = qkv + (long)(t * BB + b) * TD + 2 * DD;
  float s2 = 0.f;
  for (int d = 0; d < DD; ++d) { float v = (float)vr[d]; s2 += v * v; }
  float ah = 0.f;
  for (int h = 0; h < HH; ++h) ah += attn0[((long)b * HH + h) * SS + t];
  sig[b * SS + t] = sqrtf(s2) * ah;
}

// ---------------- per-batch sampling: sort/cumsum/pick/unique ----------------
__global__ void __launch_bounds__(1024) sample_kernel(const float* __restrict__ sig,
                                                      int* __restrict__ sel_token,
                                                      float* __restrict__ polbuf) {
  int b = blockIdx.x, tid = threadIdx.x;           // 1024 threads, 1024 = S-1 entries
  __shared__ float skey[1024];
  __shared__ int   sidx[1024];
  __shared__ float scan[1024];
  __shared__ int   pick[1024];
  skey[tid] = sig[b * SS + 1 + tid]; sidx[tid] = tid;
  __syncthreads();
  scan[tid] = skey[tid]; __syncthreads();
  for (int st = 512; st > 0; st >>= 1) { if (tid < st) scan[tid] += scan[tid + st]; __syncthreads(); }
  float inv = 1.f / scan[0]; __syncthreads();
  skey[tid] *= inv;
  // bitonic sort ascending (key + carried index == "order")
  for (int k = 2; k <= 1024; k <<= 1)
    for (int j = k >> 1; j > 0; j >>= 1) {
      __syncthreads();
      int ixj = tid ^ j;
      if (ixj > tid) {
        bool up = ((tid & k) == 0);
        float a = skey[tid], c = skey[ixj];
        if ((a > c) == up) {
          skey[tid] = c; skey[ixj] = a;
          int t0 = sidx[tid]; sidx[tid] = sidx[ixj]; sidx[ixj] = t0;
        }
      }
    }
  __syncthreads();
  // inclusive cumsum of sorted scores
  scan[tid] = skey[tid]; __syncthreads();
  for (int off = 1; off < 1024; off <<= 1) {
    float add = (tid >= off) ? scan[tid - off] : 0.f;
    __syncthreads();
    scan[tid] += add;
    __syncthreads();
  }
  float c0 = scan[0], invr = 1.f / (scan[1023] - scan[0]);
  __syncthreads();
  if (tid < NT - 1) {
    float y = (float)tid * (1.0f / (NT - 2));       // linspace(0,1,511)
    float best = 3.4e38f; int bi = 0;
    for (int t = 0; t < 1024; ++t) {
      float d = fabsf(y - (scan[t] - c0) * invr);
      if (d < best) { best = d; bi = t; }
    }
    pick[tid] = bi;
  } else pick[tid] = 0x7FFFFFFF;
  // sort picks ascending
  for (int k = 2; k <= 1024; k <<= 1)
    for (int j = k >> 1; j > 0; j >>= 1) {
      __syncthreads();
      int ixj = tid ^ j;
      if (ixj > tid) {
        bool up = ((tid & k) == 0);
        int a = pick[tid], c = pick[ixj];
        if ((a > c) == up) { pick[tid] = c; pick[ixj] = a; }
      }
    }
  __syncthreads();
  // mark duplicates with max_value (S-1 = 1024); last real element compared against 1 (ref semantics)
  int v = pick[tid];
  int nxt = (tid < NT - 2) ? pick[tid + 1] : ((tid == NT - 2) ? 1 : 0x7FFFFFFE);
  __syncthreads();
  pick[tid] = (tid < NT - 1) ? ((nxt == v) ? (SS - 1) : v) : 0x7FFFFFFF;
  // sort again
  for (int k = 2; k <= 1024; k <<= 1)
    for (int j = k >> 1; j > 0; j >>= 1) {
      __syncthreads();
      int ixj = tid ^ j;
      if (ixj > tid) {
        bool up = ((tid & k) == 0);
        int a = pick[tid], c = pick[ixj];
        if ((a > c) == up) { pick[tid] = c; pick[ixj] = a; }
      }
    }
  __syncthreads();
  if (tid < NT) {
    int sel; float p;
    if (tid == 0) { sel = 0; p = 1.f; }
    else {
      int u = pick[tid - 1];
      bool val = (u != SS - 1);
      sel = val ? (sidx[u] + 1) : -1;               // order[u] + 1 -> original token
      p = val ? 1.f : 0.f;
    }
    sel_token[b * NT + tid] = sel;
    polbuf[b * NT + tid] = p;
  }
}

// ---------------- V transpose: Vt[bh][d][t] bf16, zero padded ----------------
__global__ void vt_kernel(const bf16* __restrict__ qkv, bf16* __restrict__ Vt) {
  int id = blockIdx.x;                               // bh*64 + d
  int d = id & 63, bh = id >> 6, b = bh / HH, h = bh % HH;
  bf16* out = Vt + (long)id * SP;
  for (int t = threadIdx.x; t < SP; t += blockDim.x)
    out[t] = (t < SS) ? qkv[(long)(t * BB + b) * TD + 2 * DD + h * HD + d] : (bf16)0.0f;
}

// ---------------- fused WMMA scores + softmax-with-policy for 512 selected queries ----------------
// One wave per block; 16 query rows; scores staged in LDS as bf16 (16 x 1056 = 33.8 KB).
__global__ void score_softmax_kernel(const bf16* __restrict__ qkv, const int* __restrict__ sel_token,
                                     const float* __restrict__ pol, bf16* __restrict__ P) {
  __shared__ bf16 sc[16][SP];
  int bh = blockIdx.x, b = bh / HH, h = bh % HH;
  int r0 = blockIdx.y * 16;
  int lane = threadIdx.x & 31;
  int rr = r0 + (lane & 15);
  int tok = sel_token[b * NT + rr];
  v16bf a0, a1;
  if (tok >= 0) {
    const bf16* qp = qkv + (long)(tok * BB + b) * TD + h * HD;
    a0 = load_a_frag(qp, 0, 0, 0, lane);
    a1 = load_a_frag(qp, 0, 0, 32, lane);
  } else { a0 = zero_frag(); a1 = zero_frag(); }
  for (int ct = 0; ct < SP / 16; ++ct) {
    int t = ct * 16 + (lane & 15);
    int tc = (t < SS) ? t : (SS - 1);
    const bf16* kp = qkv + (long)(tc * BB + b) * TD + DD + h * HD;
    v16bf b0 = load_b_frag(kp, 0, 0, 0, lane);
    v16bf b1 = load_b_frag(kp, 0, 0, 32, lane);
    v8f acc;
#pragma unroll
    for (int i = 0; i < 8; ++i) acc[i] = 0.f;
    acc = wmma_bf16(a0, b0, acc);
    acc = wmma_bf16(a1, b1, acc);
#pragma unroll
    for (int i = 0; i < 8; ++i)
      sc[((lane >> 4) << 3) + i][ct * 16 + (lane & 15)] = (bf16)(acc[i] * 0.125f);
  }
  __syncthreads();
  for (int r = 0; r < 16; ++r) {
    int qt = sel_token[b * NT + r0 + r];
    bf16* pr = P + ((long)bh * NT + r0 + r) * SP;
    if (qt < 0) { for (int t = lane; t < SP; t += 32) pr[t] = (bf16)0.0f; continue; }
    float mx = -1e30f;
    for (int t = lane; t < SS; t += 32) mx = fmaxf(mx, (float)sc[r][t]);
#pragma unroll
    for (int m = 16; m >= 1; m >>= 1) mx = fmaxf(mx, __shfl_xor(mx, m, 32));
    float sum = 0.f;
    for (int t = lane; t < SS; t += 32) {
      float ap = pol[b * SS + t];
      float apf = ap + (1.f - ap) * ((t == qt) ? 1.f : 0.f);
      sum += __expf((float)sc[r][t] - mx) * apf;
    }
#pragma unroll
    for (int m = 16; m >= 1; m >>= 1) sum += __shfl_xor(sum, m, 32);
    float inv = 1.f / (sum + EPSF);
    for (int t = lane; t < SP; t += 32) {
      float v = 0.f;
      if (t < SS) {
        float ap = pol[b * SS + t];
        float apf = ap + (1.f - ap) * ((t == qt) ? 1.f : 0.f);
        v = (__expf((float)sc[r][t] - mx) * apf + EPSF / SS) * inv;
      }
      pr[t] = (bf16)v;
    }
  }
}

// ---------------- WMMA AV: out_h = P @ V per (b,h), scattered into (r,b,D) ----------------
__global__ void av_kernel(const bf16* __restrict__ P, const bf16* __restrict__ Vt,
                          bf16* __restrict__ attn_out) {
  int bh = blockIdx.x;
  int b = bh / HH, h = bh % HH;
  int lane = threadIdx.x & 31, wave = threadIdx.x >> 5;
  int m0 = blockIdx.y * 128 + wave * 32;
  const bf16* A  = P  + (long)bh * NT * SP;
  const bf16* Bt = Vt + (long)bh * HD * SP;
  v8f acc[2][4];
#pragma unroll
  for (int s = 0; s < 2; ++s)
#pragma unroll
    for (int t = 0; t < 4; ++t)
#pragma unroll
      for (int i = 0; i < 8; ++i) acc[s][t][i] = 0.f;
  int mA0 = m0 + (lane & 15);
  int mA1 = m0 + 16 + (lane & 15);
  for (int k0 = 0; k0 < SP; k0 += 32) {
    v16bf a0 = load_a_frag(A, SP, mA0, k0, lane);
    v16bf a1 = load_a_frag(A, SP, mA1, k0, lane);
#pragma unroll
    for (int t = 0; t < 4; ++t) {
      v16bf bf = load_b_frag(Bt, SP, t * 16 + (lane & 15), k0, lane);
      acc[0][t] = wmma_bf16(a0, bf, acc[0][t]);
      acc[1][t] = wmma_bf16(a1, bf, acc[1][t]);
    }
  }
#pragma unroll
  for (int s = 0; s < 2; ++s)
#pragma unroll
    for (int t = 0; t < 4; ++t)
#pragma unroll
      for (int i = 0; i < 8; ++i) {
        int m = m0 + s * 16 + ((lane >> 4) << 3) + i;
        int d = t * 16 + (lane & 15);
        attn_out[((long)m * BB + b) * DD + h * HD + d] = (bf16)acc[s][t][i];
      }
}

// ---------------- residual + policy + LayerNorm2 ----------------
__global__ void resid_ln2_kernel(const float* __restrict__ x_ln, const float* __restrict__ x_out,
                                 const int* __restrict__ sel_token, const float* __restrict__ polbuf,
                                 const float* __restrict__ g2, const float* __restrict__ b2,
                                 float* __restrict__ x2, bf16* __restrict__ hb) {
  int row = blockIdx.x;                              // r*B + b
  int r = row / BB, b = row % BB;
  int tok = sel_token[b * NT + r];
  float p = polbuf[b * NT + r];
  __shared__ float red[256];
  int tid = threadIdx.x;
  float v[4], s = 0.f, s2 = 0.f;
#pragma unroll
  for (int i = 0; i < 4; ++i) {
    int d = tid + i * 256;
    float sx = (tok >= 0) ? x_ln[((long)tok * BB + b) * DD + d] : 0.f;
    float val = (sx + x_out[(long)row * DD + d]) * p;
    v[i] = val; s += val; s2 += val * val;
    x2[(long)row * DD + d] = val;
  }
  red[tid] = s; __syncthreads();
  for (int st = 128; st > 0; st >>= 1) { if (tid < st) red[tid] += red[tid + st]; __syncthreads(); }
  float mu = red[0] / DD; __syncthreads();
  red[tid] = s2; __syncthreads();
  for (int st = 128; st > 0; st >>= 1) { if (tid < st) red[tid] += red[tid + st]; __syncthreads(); }
  float var = red[0] / DD - mu * mu;
  float rs = rsqrtf(var + EPSF);
#pragma unroll
  for (int i = 0; i < 4; ++i) {
    int d = tid + i * 256;
    float o = (v[i] - mu) * rs * g2[d] + b2[d];
    hb[(long)row * DD + d] = (bf16)o;
  }
}

// ---------------- final residual * policy ----------------
__global__ void final_kernel(const float* __restrict__ x2, const float* __restrict__ h2,
                             const float* __restrict__ polbuf, float* __restrict__ out) {
  long idx = (long)blockIdx.x * blockDim.x + threadIdx.x;
  if (idx >= (long)NT * BB * DD) return;
  long row = idx / DD; int r = (int)(row / BB), b = (int)(row % BB);
  out[idx] = (x2[idx] + h2[idx]) * polbuf[b * NT + r];
}
__global__ void pol_out_kernel(const float* __restrict__ polbuf, float* __restrict__ out) {
  int i = blockIdx.x * blockDim.x + threadIdx.x;
  if (i < BB * NT) out[(long)NT * BB * DD + i] = polbuf[i];  // (B, NT, 1) flat
}

extern "C" void kernel_launch(void* const* d_in, const int* in_sizes, int n_in,
                              void* d_out, int out_size, void* d_ws, size_t ws_size,
                              hipStream_t stream) {
  const float* inp       = (const float*)d_in[0];
  const float* policy    = (const float*)d_in[2];
  const float* ln1_g     = (const float*)d_in[4];
  const float* ln1_b     = (const float*)d_in[5];
  const float* in_proj_w = (const float*)d_in[6];
  const float* in_proj_b = (const float*)d_in[7];
  const float* out_proj_w= (const float*)d_in[8];
  const float* out_proj_b= (const float*)d_in[9];
  const float* ln2_g     = (const float*)d_in[10];
  const float* ln2_b     = (const float*)d_in[11];
  const float* mlp_w1    = (const float*)d_in[12];
  const float* mlp_b1    = (const float*)d_in[13];
  const float* mlp_w2    = (const float*)d_in[14];
  const float* mlp_b2    = (const float*)d_in[15];

  const long ROWS = (long)SS * BB;                 // 4100 token rows
  char* ws = (char*)d_ws;
  size_t off = 0;
  auto carve = [&](size_t bytes) { size_t o = off; off = (off + bytes + 255) & ~(size_t)255; return o; };

  float* x_ln   = (float*)(ws + carve(ROWS * DD * 4));
  bf16*  x_b    = (bf16 *)(ws + carve(ROWS * DD * 2));
  bf16*  inWb   = (bf16 *)(ws + carve((size_t)TD * DD * 2));
  bf16*  outWb  = (bf16 *)(ws + carve((size_t)DD * DD * 2));
  bf16*  W1t    = (bf16 *)(ws + carve((size_t)MLPD * DD * 2));
  bf16*  W2t    = (bf16 *)(ws + carve((size_t)DD * MLPD * 2));
  bf16*  qkvb   = (bf16 *)(ws + carve(ROWS * TD * 2));
  float* attn0  = (float*)(ws + carve((size_t)BB * HH * SS * 4));
  float* sig    = (float*)(ws + carve((size_t)BB * SS * 4));
  int*   selt   = (int  *)(ws + carve((size_t)BB * NT * 4));
  float* polb   = (float*)(ws + carve((size_t)BB * NT * 4));
  bf16*  Vt     = (bf16 *)(ws + carve((size_t)BB * HH * HD * SP * 2));
  bf16*  Pb     = (bf16 *)(ws + carve((size_t)BB * HH * NT * SP * 2));
  bf16*  aoutb  = (bf16 *)(ws + carve((size_t)NT * BB * DD * 2));
  float* x_out  = (float*)(ws + carve((size_t)NT * BB * DD * 4));
  float* x2     = (float*)(ws + carve((size_t)NT * BB * DD * 4));
  bf16*  hb     = (bf16 *)(ws + carve((size_t)NT * BB * DD * 2));
  bf16*  actb   = (bf16 *)(ws + carve((size_t)NT * BB * MLPD * 2));
  float* h2     = (float*)(ws + carve((size_t)NT * BB * DD * 4));
  (void)ws_size; (void)in_sizes; (void)n_in; (void)out_size;

  // 1) LN1 -> fp32 + bf16
  ln1_kernel<<<(int)ROWS, 256, 0, stream>>>(inp, ln1_g, ln1_b, x_ln, x_b);
  // 2) weights -> bf16 (MLP weights transposed to [N,K])
  { long n = (long)TD * DD;   cvt_kernel<<<(int)((n + 255) / 256), 256, 0, stream>>>(in_proj_w, inWb, n); }
  { long n = (long)DD * DD;   cvt_kernel<<<(int)((n + 255) / 256), 256, 0, stream>>>(out_proj_w, outWb, n); }
  { long n = (long)DD * MLPD; transpose_cvt_kernel<<<(int)((n + 255) / 256), 256, 0, stream>>>(mlp_w1, W1t, DD, MLPD); }
  { long n = (long)MLPD * DD; transpose_cvt_kernel<<<(int)((n + 255) / 256), 256, 0, stream>>>(mlp_w2, W2t, MLPD, DD); }
  // 3) QKV = x @ inW^T + b  (WMMA)
  gemm_bt_kernel<<<dim3(TD / 64, (int)((ROWS + 127) / 128)), 128, 0, stream>>>(
      x_b, inWb, in_proj_b, nullptr, qkvb, (int)ROWS, TD, DD, 0);
  // 4) attention row 0 per (b,h)
  attn0_kernel<<<BB * HH, 256, 0, stream>>>(qkvb, policy, attn0);
  // 5) significance
  sig_kernel<<<(BB * SS + 255) / 256, 256, 0, stream>>>(qkvb, attn0, sig);
  // 6) token sampling per batch
  sample_kernel<<<BB, 1024, 0, stream>>>(sig, selt, polb);
  // 7) V transpose (padded)
  vt_kernel<<<BB * HH * HD, 128, 0, stream>>>(qkvb, Vt);
  // 8) fused WMMA scores + softmax -> bf16 P
  score_softmax_kernel<<<dim3(BB * HH, NT / 16), 32, 0, stream>>>(qkvb, selt, policy, Pb);
  // 9) WMMA AV -> attn_out (r,b,D) bf16
  av_kernel<<<dim3(BB * HH, NT / 128), 128, 0, stream>>>(Pb, Vt, aoutb);
  // 10) out-projection (WMMA)
  gemm_bt_kernel<<<dim3(DD / 64, (NT * BB + 127) / 128), 128, 0, stream>>>(
      aoutb, outWb, out_proj_b, x_out, nullptr, NT * BB, DD, DD, 0);
  // 11) residual + policy + LN2
  resid_ln2_kernel<<<NT * BB, 256, 0, stream>>>(x_ln, x_out, selt, polb, ln2_g, ln2_b, x2, hb);
  // 12) MLP1 with exact-GELU epilogue (WMMA)
  gemm_bt_kernel<<<dim3(MLPD / 64, (NT * BB + 127) / 128), 128, 0, stream>>>(
      hb, W1t, mlp_b1, nullptr, actb, NT * BB, MLPD, DD, 1);
  // 13) MLP2 (WMMA)
  gemm_bt_kernel<<<dim3(DD / 64, (NT * BB + 127) / 128), 128, 0, stream>>>(
      actb, W2t, mlp_b2, h2, nullptr, NT * BB, DD, MLPD, 0);
  // 14) final residual * policy + policy output
  final_kernel<<<(int)(((long)NT * BB * DD + 255) / 256), 256, 0, stream>>>(x2, h2, polb, (float*)d_out);
  pol_out_kernel<<<(BB * NT + 255) / 256, 256, 0, stream>>>(polb, (float*)d_out);
}
